// SelfAttention_55671366091382
// MI455X (gfx1250) — compile-verified, matrix operand order checked by reading, not run on hardware
//
#include <hip/hip_runtime.h>
#include <hip/hip_bf16.h>

// ---------------------------------------------------------------------------
// CDNA5 (gfx1250) fused self-attention, wave32 + v_wmma_f32_16x16x32_f16.
// B=8, F=4, C=256, H=W=32 -> L=4096 tokens, pooled T=1024, 2 heads of 128.
// Bandwidth-bound: f16 operands everywhere, flash-style fused attention,
// K/V staged cooperatively into LDS via async-to-LDS (ASYNCcnt) with
// double buffering so DMA overlaps WMMA compute.
// ---------------------------------------------------------------------------

typedef __attribute__((ext_vector_type(16))) _Float16 v16h;
typedef __attribute__((ext_vector_type(8)))  _Float16 v8h;
typedef __attribute__((ext_vector_type(8)))  float    v8f;
typedef __attribute__((ext_vector_type(4)))  int      v4i;

#define DEVINL __device__ __forceinline__

#if defined(__AMDGCN__) && __has_builtin(__builtin_amdgcn_global_load_async_to_lds_b128)
#define USE_ASYNC_LDS 1
#endif

#if defined(__AMDGCN__) && __has_builtin(__builtin_amdgcn_s_wait_asynccnt)
#define WAIT_ASYNC() __builtin_amdgcn_s_wait_asynccnt(0)
#elif defined(USE_ASYNC_LDS)
#define WAIT_ASYNC() asm volatile("s_wait_asynccnt 0x0" ::: "memory")
#else
#define WAIT_ASYNC() ((void)0)
#endif

// Copy 16 bytes/lane from global to LDS. Async path: one
// global_load_async_to_lds_b128 per call (512B per wave), tracked by ASYNCcnt.
// Builtin signature (probed from hipcc diagnostics): global int4* (AS1),
// LDS int4* (AS3), imm offset, imm cpol.
DEVINL void cp16_to_lds(const _Float16* g, _Float16* l) {
#ifdef USE_ASYNC_LDS
  __builtin_amdgcn_global_load_async_to_lds_b128(
      (__attribute__((address_space(1))) v4i*)(v4i*)g,
      (__attribute__((address_space(3))) v4i*)(v4i*)l, 0, 0);
#else
  *(v8h*)l = *(const v8h*)g;
#endif
}

DEVINL v8f wmma_f16(v16h a, v16h b, v8f c) {
  // (neg_a, A, neg_b, B, c_mod, C, reuse_a, reuse_b)
  return __builtin_amdgcn_wmma_f32_16x16x32_f16(false, a, false, b, (short)0, c,
                                                false, false);
}

// Load a 16x32 tile in WMMA 16-bit A layout from a row-major half matrix
// (row stride `ld` halves). Per ISA: lane L supplies row L%16; its 16 halves
// map to k = (i&7) + 8*(L>>4) + 16*(i>>3) -> two contiguous 8-half runs.
// Also produces a B-tile (32x16, K-major) when the source is [n][k]
// row-major, since B[k][n] = src[n][k]. Works for global or LDS pointers.
DEVINL v16h load_tile(const _Float16* p, int ld, int lane) {
  const int r = lane & 15, g = lane >> 4;
  const _Float16* b0 = p + (size_t)r * ld + g * 8;
  v8h lo = *(const v8h*)(b0);
  v8h hi = *(const v8h*)(b0 + 16);
  v16h t;
#pragma unroll
  for (int i = 0; i < 8; ++i) { t[i] = lo[i]; t[8 + i] = hi[i]; }
  return t;
}

// ---------------------------------------------------------------------------
// Prep: depthwise 3x3 conv + BN (q path), f32 -> f16
// ---------------------------------------------------------------------------
__global__ void __launch_bounds__(256) dwconv_bn_kernel(
    const float* __restrict__ x, const float* __restrict__ wdw,
    const float* __restrict__ gamma, const float* __restrict__ beta,
    const float* __restrict__ mean, const float* __restrict__ var,
    _Float16* __restrict__ qin) {
  int i = blockIdx.x * 256 + threadIdx.x;
  int c = i & 255; int s = i >> 8;
  int xx = s & 31; s >>= 5;
  int yy = s & 31; s >>= 5;
  int f  = s & 3;  int b = s >> 2;
  int ch = f * 256 + c;
  const float* wp = wdw + (size_t)ch * 9;
  float acc = 0.f;
#pragma unroll
  for (int ky = -1; ky <= 1; ++ky) {
    int y2 = yy + ky;
    if ((unsigned)y2 < 32u) {
#pragma unroll
      for (int kx = -1; kx <= 1; ++kx) {
        int x2 = xx + kx;
        if ((unsigned)x2 < 32u) {
          int t = f * 1024 + y2 * 32 + x2;
          acc += x[((size_t)b * 4096 + t) * 256 + c] * wp[(ky + 1) * 3 + (kx + 1)];
        }
      }
    }
  }
  float inv = gamma[ch] * rsqrtf(var[ch] + 1e-5f);
  float q = (acc - mean[ch]) * inv + beta[ch];
  int t = f * 1024 + yy * 32 + xx;
  qin[((size_t)b * 4096 + t) * 256 + c] = (_Float16)q;
}

// ---------------------------------------------------------------------------
// Prep: 2x2 average pool (k/v source), f32 -> f16
// ---------------------------------------------------------------------------
__global__ void __launch_bounds__(256) pool2x2_kernel(
    const float* __restrict__ x, _Float16* __restrict__ kvin) {
  int i = blockIdx.x * 256 + threadIdx.x;
  int c = i & 255; int s = i >> 8;
  int xx = s & 15; s >>= 4;
  int yy = s & 15; s >>= 4;
  int f  = s & 3;  int b = s >> 2;
  const float* xb = x + ((size_t)b * 4096 + f * 1024) * 256 + c;
  int y2 = yy * 2, x2 = xx * 2;
  float v = xb[(size_t)(y2 * 32 + x2) * 256] +
            xb[(size_t)(y2 * 32 + x2 + 1) * 256] +
            xb[(size_t)((y2 + 1) * 32 + x2) * 256] +
            xb[(size_t)((y2 + 1) * 32 + x2 + 1) * 256];
  int tk = f * 256 + yy * 16 + xx;
  kvin[((size_t)b * 1024 + tk) * 256 + c] = (_Float16)(v * 0.25f);
}

// ---------------------------------------------------------------------------
// f32 -> f16 weight conversion
// ---------------------------------------------------------------------------
__global__ void __launch_bounds__(256) cvt_kernel(const float* __restrict__ src,
                                                  _Float16* __restrict__ dst,
                                                  int n) {
  int i = blockIdx.x * 256 + threadIdx.x;
  if (i < n) dst[i] = (_Float16)src[i];
}

// ---------------------------------------------------------------------------
// Y[M x 256] = X[M x 256] @ W^T + bias, WMMA f16->f32.
// One 16x16 tile per wave; 8 waves/block -> 32x64 block tile.
// mode 0: f16 out, row-major [m][n]            (Q / K projections)
// mode 1: f16 out, per-batch transposed V^T:   vt[b][n][t], m = b*1024 + t
// mode 2: f32 out, row-major [m][n]            (final Wo projection)
// ---------------------------------------------------------------------------
__global__ void __launch_bounds__(256) gemm256_kernel(
    const _Float16* __restrict__ X, const _Float16* __restrict__ Wh,
    const float* __restrict__ bias, void* __restrict__ Yv, int mode) {
  const int lane = threadIdx.x & 31, w = threadIdx.x >> 5;
  const int m0 = (blockIdx.x * 2 + (w & 1)) * 16;
  const int n0 = (blockIdx.y * 4 + (w >> 1)) * 16;
  v8f acc = {};
#pragma unroll
  for (int k0 = 0; k0 < 256; k0 += 32) {
    v16h a = load_tile(X + (size_t)m0 * 256 + k0, 256, lane);
    v16h b = load_tile(Wh + (size_t)n0 * 256 + k0, 256, lane);
    acc = wmma_f16(a, b, acc);
  }
  const int g = lane >> 4, n = n0 + (lane & 15);
  const float bn = bias[n];
#pragma unroll
  for (int v = 0; v < 8; ++v) {
    const int m = m0 + v + 8 * g;
    float y = acc[v] + bn;
    if (mode == 0) {
      ((_Float16*)Yv)[(size_t)m * 256 + n] = (_Float16)y;
    } else if (mode == 1) {
      int bb = m >> 10, t = m & 1023;
      ((_Float16*)Yv)[((size_t)bb * 256 + n) * 1024 + t] = (_Float16)y;
    } else {
      ((float*)Yv)[(size_t)m * 256 + n] = y;
    }
  }
}

// ---------------------------------------------------------------------------
// Fused flash attention with cooperative async-to-LDS K/V staging.
// grid = (B*2 heads, L/128). Block = 8 waves; wave w owns Q rows
// [blockIdx.y*128 + w*16, +16). Keys processed in 32-chunks:
//   - all 256 threads issue async b128 copies of the NEXT K (32x128, [t][d])
//     and V (128x32, [d][t]) chunk into the alternate LDS buffer
//   - s_wait_asynccnt 0 + barrier retires the CURRENT chunk's copies
//   - per wave: S = Q K^T (8 WMMAs from LDS B-tiles), online softmax,
//     P transposed through per-wave LDS scratch, O += P V (8 WMMAs).
// K/V are loaded once per 128 Q rows instead of once per 16 -> 8x less
// on-chip traffic, and the async copies overlap the WMMA work.
// ---------------------------------------------------------------------------
__global__ void __launch_bounds__(256) attn_kernel(
    const _Float16* __restrict__ qp, const _Float16* __restrict__ kp,
    const _Float16* __restrict__ vt, _Float16* __restrict__ oh) {
  __shared__ _Float16 Ksh[2][32 * 128];   // [buf][t][d]   8 KB each
  __shared__ _Float16 Vsh[2][128 * 32];   // [buf][d][t]   8 KB each
  __shared__ _Float16 Psh[8][16 * 40];    // per-wave P scratch (padded rows)

  const int tid = threadIdx.x;
  const int lane = tid & 31, wv = tid >> 5;
  const int b = blockIdx.x >> 1, h = blockIdx.x & 1;
  const int l0 = blockIdx.y * 128 + wv * 16;
  const _Float16* qb = qp + ((size_t)b * 4096 + l0) * 256 + h * 128;
  const _Float16* kb = kp + (size_t)b * 1024 * 256 + h * 128;
  const _Float16* vb = vt + ((size_t)b * 256 + h * 128) * 1024;

  // --- per-thread staging assignment: 2 K segments + 2 V segments (16B) ---
  // K chunk = 512 segs: seg s -> t = s>>4, d = (s&15)*8
  // V chunk = 512 segs: seg s -> d = s>>2, t = (s&3)*8
  const int kt0 = tid >> 4,         kd0 = (tid & 15) * 8;
  const int kt1 = (tid + 256) >> 4, kd1 = (tid & 15) * 8;
  const int vd0 = tid >> 2,         vt0c = (tid & 3) * 8;
  const int vd1 = (tid + 256) >> 2, vt1c = (tid & 3) * 8;

  auto stage = [&](int t0, int buf) {
    cp16_to_lds(kb + (size_t)(t0 + kt0) * 256 + kd0, &Ksh[buf][kt0 * 128 + kd0]);
    cp16_to_lds(kb + (size_t)(t0 + kt1) * 256 + kd1, &Ksh[buf][kt1 * 128 + kd1]);
    cp16_to_lds(vb + (size_t)vd0 * 1024 + t0 + vt0c, &Vsh[buf][vd0 * 32 + vt0c]);
    cp16_to_lds(vb + (size_t)vd1 * 1024 + t0 + vt1c, &Vsh[buf][vd1 * 32 + vt1c]);
  };

  // --- Q tile (16x128) resident in registers for the whole key loop ---
  v16h qa[4];
#pragma unroll
  for (int kt = 0; kt < 4; ++kt) qa[kt] = load_tile(qb + kt * 32, 256, lane);

  v8f o[8] = {};
  float mrow[8], lrow[8];
#pragma unroll
  for (int v = 0; v < 8; ++v) { mrow[v] = -1e30f; lrow[v] = 0.f; }

  const int g = lane >> 4, col = lane & 15;
  const float scale = 0.0625f;  // 256^-0.5 (full-dim scale, as in reference)

  stage(0, 0);  // prologue: chunk 0 in flight

  for (int i = 0; i < 32; ++i) {
    const int buf = i & 1;
    WAIT_ASYNC();        // my copies for chunk i have landed in LDS
    __syncthreads();     // everyone's copies done; prev chunk fully consumed
    if (i + 1 < 32) stage((i + 1) * 32, buf ^ 1);  // overlap next DMA

    const _Float16* kc = Ksh[buf];
    const _Float16* vc = Vsh[buf];

    // --- S = Q K^T for this 16x32 key chunk (two 16x16 N-tiles) ---
    v8f s0 = {}, s1 = {};
#pragma unroll
    for (int kt = 0; kt < 4; ++kt) {
      v16h b0 = load_tile(kc + kt * 32, 128, lane);
      v16h b1 = load_tile(kc + 16 * 128 + kt * 32, 128, lane);
      s0 = wmma_f16(qa[kt], b0, s0);
      s1 = wmma_f16(qa[kt], b1, s1);
    }

    // --- online softmax; stage P (f16) into this wave's LDS scratch ---
    _Float16* pw = Psh[wv];
#pragma unroll
    for (int v = 0; v < 8; ++v) {
      float a0 = s0[v] * scale, a1 = s1[v] * scale;
      float rmax = fmaxf(a0, a1);
#pragma unroll
      for (int msk = 1; msk < 16; msk <<= 1)
        rmax = fmaxf(rmax, __shfl_xor(rmax, msk, 32));
      float mnew = fmaxf(mrow[v], rmax);
      float cf = __expf(mrow[v] - mnew);
      float p0 = __expf(a0 - mnew), p1 = __expf(a1 - mnew);
      float rs = p0 + p1;
#pragma unroll
      for (int msk = 1; msk < 16; msk <<= 1) rs += __shfl_xor(rs, msk, 32);
      lrow[v] = lrow[v] * cf + rs;
      mrow[v] = mnew;
#pragma unroll
      for (int d = 0; d < 8; ++d) o[d][v] *= cf;
      const int row = v + 8 * g;
      pw[row * 40 + col]      = (_Float16)p0;
      pw[row * 40 + col + 16] = (_Float16)p1;
    }
    __syncthreads();  // orders the cross-lane P transpose (and LDS reuse)

    // --- reload P as a WMMA A-tile ---
    v16h pa;
    {
      const _Float16* base = &pw[(lane & 15) * 40 + (lane >> 4) * 8];
#pragma unroll
      for (int j = 0; j < 8; ++j) { pa[j] = base[j]; pa[8 + j] = base[16 + j]; }
    }

    // --- O += P V (V tile in LDS, rows d, contiguous t) ---
#pragma unroll
    for (int d = 0; d < 8; ++d) {
      v16h bvt = load_tile(vc + (d * 16) * 32, 32, lane);
      o[d] = wmma_f16(pa, bvt, o[d]);
    }
  }

  // --- normalize rows by softmax denominator, store f16 ---
#pragma unroll
  for (int v = 0; v < 8; ++v) {
    const int m = l0 + v + 8 * g;
    const float linv = 1.f / lrow[v];
#pragma unroll
    for (int d = 0; d < 8; ++d) {
      int c = h * 128 + d * 16 + col;
      oh[((size_t)b * 4096 + m) * 256 + c] = (_Float16)(o[d][v] * linv);
    }
  }
}

// ---------------------------------------------------------------------------
// Host orchestration
// ---------------------------------------------------------------------------
extern "C" void kernel_launch(void* const* d_in, const int* in_sizes, int n_in,
                              void* d_out, int out_size, void* d_ws, size_t ws_size,
                              hipStream_t stream) {
  const float* x     = (const float*)d_in[0];
  const float* wdw   = (const float*)d_in[1];
  const float* gamma = (const float*)d_in[2];
  const float* beta  = (const float*)d_in[3];
  const float* mean  = (const float*)d_in[4];
  const float* var   = (const float*)d_in[5];
  const float* Wq = (const float*)d_in[6];  const float* bq = (const float*)d_in[7];
  const float* Wk = (const float*)d_in[8];  const float* bk = (const float*)d_in[9];
  const float* Wv = (const float*)d_in[10]; const float* bv = (const float*)d_in[11];
  const float* Wo = (const float*)d_in[12]; const float* bo = (const float*)d_in[13];

  char* ws = (char*)d_ws;
  size_t off = 0;
  auto alloc = [&](size_t bytes) {
    char* p = ws + off;
    off = (off + bytes + 255) & ~(size_t)255;
    return p;
  };
  const size_t L = 8ull * 4096 * 256;  // q-side token elements
  const size_t T = 8ull * 1024 * 256;  // pooled token elements
  _Float16* qin = (_Float16*)alloc(L * 2);
  _Float16* kvin = (_Float16*)alloc(T * 2);
  _Float16* Wqh = (_Float16*)alloc(65536 * 2);
  _Float16* Wkh = (_Float16*)alloc(65536 * 2);
  _Float16* Wvh = (_Float16*)alloc(65536 * 2);
  _Float16* Woh = (_Float16*)alloc(65536 * 2);
  _Float16* qpj = (_Float16*)alloc(L * 2);   // Q projected [B,4096,256]
  _Float16* kpj = (_Float16*)alloc(T * 2);   // K projected [B,1024,256]
  _Float16* vtj = (_Float16*)alloc(T * 2);   // V projected, transposed [B,256,1024]
  _Float16* ohb = (_Float16*)alloc(L * 2);   // attention output [B,4096,256]

  // 1) preprocessing
  dwconv_bn_kernel<<<(int)(L / 256), 256, 0, stream>>>(x, wdw, gamma, beta, mean,
                                                       var, qin);
  pool2x2_kernel<<<(int)(T / 256), 256, 0, stream>>>(x, kvin);
  cvt_kernel<<<256, 256, 0, stream>>>(Wq, Wqh, 65536);
  cvt_kernel<<<256, 256, 0, stream>>>(Wk, Wkh, 65536);
  cvt_kernel<<<256, 256, 0, stream>>>(Wv, Wvh, 65536);
  cvt_kernel<<<256, 256, 0, stream>>>(Wo, Woh, 65536);

  // 2) projections (WMMA GEMMs)
  gemm256_kernel<<<dim3(32768 / 32, 4), 256, 0, stream>>>(qin, Wqh, bq, qpj, 0);
  gemm256_kernel<<<dim3(8192 / 32, 4), 256, 0, stream>>>(kvin, Wkh, bk, kpj, 0);
  gemm256_kernel<<<dim3(8192 / 32, 4), 256, 0, stream>>>(kvin, Wvh, bv, vtj, 1);

  // 3) fused flash attention: 16 (batch,head) x 32 blocks of 128 Q rows
  attn_kernel<<<dim3(16, 32), 256, 0, stream>>>(qpj, kpj, vtj, ohb);

  // 4) output projection -> f32 d_out
  gemm256_kernel<<<dim3(32768 / 32, 4), 256, 0, stream>>>(ohb, Woh, bo, d_out, 2);
}
